// SimpleTreeModel_67937792688559
// MI455X (gfx1250) — matrix operand order repaired and checked by reading, not compiled, not run on hardware
//
#include <hip/hip_runtime.h>
#include <stdint.h>

#define N_NODES   100000
#define N_EDGES   1600000
#define F_IN      64
#define HID       128
#define HEAD_H    256
#define OUT_F     32
#define N_GRAPHS  512
#define NEG_SLOPE 0.01f

typedef __attribute__((ext_vector_type(16))) __bf16 v16bf;
typedef __attribute__((ext_vector_type(8)))  float  v8f;

// ---------- helpers ----------
__device__ __forceinline__ uint32_t f2bf(float f) {
  uint32_t u = __builtin_bit_cast(uint32_t, f);
  u += 0x7FFFu + ((u >> 16) & 1u);   // round-to-nearest-even
  return u >> 16;
}
__device__ __forceinline__ uint32_t pack_bf2(float lo, float hi) {
  return f2bf(lo) | (f2bf(hi) << 16);
}
__device__ __forceinline__ void atomic_add_f32(float* p, float v) {
  __hip_atomic_fetch_add(p, v, __ATOMIC_RELAXED, __HIP_MEMORY_SCOPE_AGENT);
}

// ---------- zero fill ----------
__global__ void zero_f32_kernel(float* __restrict__ p, int n) {
  int i = blockIdx.x * blockDim.x + threadIdx.x;
  if (i < n) p[i] = 0.0f;
}

// ---------- pack fp32 weights [K][N] into WMMA-fragment-ordered bf16 pairs ----
// Output dword index o = ((kbIdx*nNt + nt)*32 + lane)*8 + j
//   lane=(r,h): value = pack(W[2*k2][col], W[2*k2+1][col]),
//   k2 = kbIdx*16 + 8h + j, col = nt*16 + r   (B fragment for 16x16x32 bf16)
__global__ void pack_weights_swz_kernel(const float* __restrict__ w,
                                        uint32_t* __restrict__ wp,
                                        int K, int N, int ntShift /* log2(N/16) */) {
  int o = blockIdx.x * blockDim.x + threadIdx.x;
  int total = (K >> 1) * N;          // dwords
  if (o >= total) return;
  int j     = o & 7;
  int lane  = (o >> 3) & 31;
  int nt    = (o >> 8) & ((1 << ntShift) - 1);
  int kbIdx = o >> (8 + ntShift);
  int r = lane & 15, h = lane >> 4;
  int k2  = (kbIdx << 4) + (h << 3) + j;
  int col = (nt << 4) + r;
  wp[o] = pack_bf2(w[(2 * k2) * N + col], w[(2 * k2 + 1) * N + col]);
}

// ---------- fused edge message + scatter-add ----------
// one thread = 4 consecutive features of one edge:
//   m = relu(x[src] + ea0*We[0] + ea1*We[1] + be); atomicAdd into agg[dst]
__global__ void __launch_bounds__(256)
edge_msg_kernel(const float* __restrict__ x,
                const int*   __restrict__ srcIdx,
                const int*   __restrict__ dstIdx,
                const float* __restrict__ ea,
                const float* __restrict__ We,
                const float* __restrict__ be,
                float*       __restrict__ agg,
                int D, int cShift /* log2(D/4) */) {
  int t = blockIdx.x * blockDim.x + threadIdx.x;
  int nChunk = N_EDGES << cShift;
  if (t >= nChunk) return;
  int e = t >> cShift;
  int c = (t & ((1 << cShift) - 1)) << 2;
  int s = srcIdx[e];
  int d = dstIdx[e];
  float a0 = ea[2 * e];
  float a1 = ea[2 * e + 1];
  float4 w0 = *(const float4*)(We + c);
  float4 w1 = *(const float4*)(We + D + c);
  float4 b4 = *(const float4*)(be + c);
  float4 xv = *(const float4*)(x + (long)s * D + c);
  float m0 = fmaxf(xv.x + a0 * w0.x + a1 * w1.x + b4.x, 0.0f);
  float m1 = fmaxf(xv.y + a0 * w0.y + a1 * w1.y + b4.y, 0.0f);
  float m2 = fmaxf(xv.z + a0 * w0.z + a1 * w1.z + b4.z, 0.0f);
  float m3 = fmaxf(xv.w + a0 * w0.w + a1 * w1.w + b4.w, 0.0f);
  float* ap = agg + (long)d * D + c;
  atomic_add_f32(ap + 0, m0);
  atomic_add_f32(ap + 1, m1);
  atomic_add_f32(ap + 2, m2);
  atomic_add_f32(ap + 3, m3);
}

// ---------- WMMA GEMM: out[M,N] = act((A1 (+A2)) @ W + bias) ----------
// Templated on (K, N) so all B strides fold into load immediate offsets.
// Block = 256 thr = 8 waves; block owns 16 rows; wave owns a 16-col tile.
// A staged in LDS in fragment order -> 2x ds_load_b128 per WMMA.
// Bs pre-swizzled weight fragments -> 2x global_load_b128 (immediate offsets).
// Two interleaved accumulators break the WMMA->WMMA dependency chain.
template <int K, int N>
__global__ void __launch_bounds__(256)
gemm_wmma_bf16_kernel(const float*    __restrict__ A1,
                      const float*    __restrict__ A2,   // nullable
                      const uint32_t* __restrict__ Bs,   // fragment-ordered
                      const float*    __restrict__ bias,
                      float*          __restrict__ outp,
                      int act) {
  constexpr int halfK  = K / 2;
  constexpr int nKb    = K / 32;
  constexpr int nTiles = N / 16;
  __shared__ uint32_t sA[nKb * 256];
  const int tid  = threadIdx.x;
  const int row0 = blockIdx.x << 4;

  // cooperative: sum A1(+A2), convert to packed bf16, scatter into fragment
  // order. Row r, dword dw: kbIdx = dw/16; local dword ld maps to lane-half
  // hh = ld>>3 and fragment slot jj = ld&7.
  for (int i = tid; i < 16 * halfK; i += 256) {
    int r  = i / halfK;
    int dw = i % halfK;
    long base = (long)(row0 + r) * K + 2 * dw;
    float lo = A1[base];
    float hi = A1[base + 1];
    if (A2 != nullptr) { lo += A2[base]; hi += A2[base + 1]; }
    int kbIdx = dw >> 4;
    int ld = dw & 15;
    int hh = ld >> 3;
    int jj = ld & 7;
    sA[(kbIdx << 8) + ((r + (hh << 4)) << 3) + jj] = pack_bf2(lo, hi);
  }
  __syncthreads();

  const int lane = tid & 31;
  const int wave = tid >> 5;
  const int r = lane & 15;
  const int h = lane >> 4;

  for (int nt = wave; nt < nTiles; nt += 8) {       // uniform per wave
    const uint32_t* bbase = Bs + (size_t)(nt * 32 + lane) * 8;
    const uint32_t* abase = sA + lane * 8;
    v8f acc0 = {};
    v8f acc1 = {};
#pragma unroll
    for (int kb = 0; kb < nKb; ++kb) {
      union { uint32_t u[8]; uint4 q[2]; v16bf v; } af, bfr;
      const uint4* ap = (const uint4*)(abase + kb * 256);
      af.q[0] = ap[0];
      af.q[1] = ap[1];
      const uint4* bp = (const uint4*)(bbase + (size_t)kb * nTiles * 256);
      bfr.q[0] = bp[0];
      bfr.q[1] = bp[1];
      if (kb & 1)
        acc1 = __builtin_amdgcn_wmma_f32_16x16x32_bf16(false, af.v, false, bfr.v,
                                                       (short)0, acc1, false, false);
      else
        acc0 = __builtin_amdgcn_wmma_f32_16x16x32_bf16(false, af.v, false, bfr.v,
                                                       (short)0, acc0, false, false);
    }
    // C/D layout: vgpr j -> row = row0 + j + 8h, col = nt*16 + r
    const int col = nt * 16 + r;
    const float bv = bias[col];
#pragma unroll
    for (int j = 0; j < 8; ++j) {
      float o = acc0[j] + acc1[j] + bv;
      if (act) o = (o >= 0.0f) ? o : NEG_SLOPE * o;
      outp[(size_t)(row0 + j + (h << 3)) * N + col] = o;
    }
  }
}

// ---------- graph pooling: pooled[g] += h[node] ----------
__global__ void __launch_bounds__(256)
pool_kernel(const float* __restrict__ hfeat,
            const int*   __restrict__ batch,
            float*       __restrict__ pooled) {
  int t = blockIdx.x * blockDim.x + threadIdx.x;
  int total = N_NODES * (HID / 4);
  if (t >= total) return;
  int node = t >> 5;
  int c = (t & 31) << 2;
  int g = batch[node];
  float4 v = *(const float4*)(hfeat + (long)node * HID + c);
  float* pp = pooled + (long)g * HID + c;
  atomic_add_f32(pp + 0, v.x);
  atomic_add_f32(pp + 1, v.y);
  atomic_add_f32(pp + 2, v.z);
  atomic_add_f32(pp + 3, v.w);
}

static inline size_t align256(size_t x) { return (x + 255) & ~(size_t)255; }
static inline int cdiv(long a, int b) { return (int)((a + b - 1) / b); }

extern "C" void kernel_launch(void* const* d_in, const int* in_sizes, int n_in,
                              void* d_out, int out_size, void* d_ws, size_t ws_size,
                              hipStream_t stream) {
  const float* x     = (const float*)d_in[0];
  const int*   ei    = (const int*)  d_in[1];
  const float* ea    = (const float*)d_in[2];
  const int*   batch = (const int*)  d_in[3];
  const float* We0 = (const float*)d_in[4];
  const float* be0 = (const float*)d_in[5];
  const float* Wn0 = (const float*)d_in[6];
  const float* bn0 = (const float*)d_in[7];
  const float* We1 = (const float*)d_in[8];
  const float* be1 = (const float*)d_in[9];
  const float* Wn1 = (const float*)d_in[10];
  const float* bn1 = (const float*)d_in[11];
  const float* We2 = (const float*)d_in[12];
  const float* be2 = (const float*)d_in[13];
  const float* Wn2 = (const float*)d_in[14];
  const float* bn2 = (const float*)d_in[15];
  const float* Wh0 = (const float*)d_in[16];
  const float* bh0 = (const float*)d_in[17];
  const float* Wh1 = (const float*)d_in[18];
  const float* bh1 = (const float*)d_in[19];
  float* out = (float*)d_out;
  const int* src = ei;
  const int* dst = ei + N_EDGES;

  // workspace carve-up (~155 MB)
  char* w = (char*)d_ws;
  float* bufA   = (float*)w; w += align256((size_t)N_NODES * HID * sizeof(float));
  float* bufB   = (float*)w; w += align256((size_t)N_NODES * HID * sizeof(float));
  float* bufC   = (float*)w; w += align256((size_t)N_NODES * HID * sizeof(float));
  float* pooled = (float*)w; w += align256((size_t)N_GRAPHS * HID * sizeof(float));
  float* hidden = (float*)w; w += align256((size_t)N_GRAPHS * HEAD_H * sizeof(float));
  uint32_t* Wp0  = (uint32_t*)w; w += align256((size_t)(F_IN / 2) * HID * 4);
  uint32_t* Wp1  = (uint32_t*)w; w += align256((size_t)(HID / 2) * HID * 4);
  uint32_t* Wp2  = (uint32_t*)w; w += align256((size_t)(HID / 2) * HID * 4);
  uint32_t* Wph0 = (uint32_t*)w; w += align256((size_t)(HID / 2) * HEAD_H * 4);
  uint32_t* Wph1 = (uint32_t*)w; w += align256((size_t)(HEAD_H / 2) * OUT_F * 4);

  // pack all dense weights into fragment-ordered bf16 pairs
  pack_weights_swz_kernel<<<cdiv((F_IN / 2) * HID, 256), 256, 0, stream>>>(Wn0, Wp0, F_IN, HID, 3);
  pack_weights_swz_kernel<<<cdiv((HID / 2) * HID, 256), 256, 0, stream>>>(Wn1, Wp1, HID, HID, 3);
  pack_weights_swz_kernel<<<cdiv((HID / 2) * HID, 256), 256, 0, stream>>>(Wn2, Wp2, HID, HID, 3);
  pack_weights_swz_kernel<<<cdiv((HID / 2) * HEAD_H, 256), 256, 0, stream>>>(Wh0, Wph0, HID, HEAD_H, 4);
  pack_weights_swz_kernel<<<cdiv((HEAD_H / 2) * OUT_F, 256), 256, 0, stream>>>(Wh1, Wph1, HEAD_H, OUT_F, 1);

  // ---- layer 0 (64 -> 128) ----
  zero_f32_kernel<<<cdiv((long)N_NODES * F_IN, 256), 256, 0, stream>>>(bufA, N_NODES * F_IN);
  edge_msg_kernel<<<cdiv((long)N_EDGES * (F_IN / 4), 256), 256, 0, stream>>>(
      x, src, dst, ea, We0, be0, bufA, F_IN, 4);
  gemm_wmma_bf16_kernel<F_IN, HID><<<N_NODES / 16, 256, 0, stream>>>(
      bufA, x, Wp0, bn0, bufB, 1);

  // ---- layer 1 (128 -> 128) ----
  zero_f32_kernel<<<cdiv((long)N_NODES * HID, 256), 256, 0, stream>>>(bufA, N_NODES * HID);
  edge_msg_kernel<<<cdiv((long)N_EDGES * (HID / 4), 256), 256, 0, stream>>>(
      bufB, src, dst, ea, We1, be1, bufA, HID, 5);
  gemm_wmma_bf16_kernel<HID, HID><<<N_NODES / 16, 256, 0, stream>>>(
      bufA, bufB, Wp1, bn1, bufC, 1);

  // ---- layer 2 (128 -> 128) ----
  zero_f32_kernel<<<cdiv((long)N_NODES * HID, 256), 256, 0, stream>>>(bufA, N_NODES * HID);
  edge_msg_kernel<<<cdiv((long)N_EDGES * (HID / 4), 256), 256, 0, stream>>>(
      bufC, src, dst, ea, We2, be2, bufA, HID, 5);
  gemm_wmma_bf16_kernel<HID, HID><<<N_NODES / 16, 256, 0, stream>>>(
      bufA, bufC, Wp2, bn2, bufB, 1);

  // ---- graph pooling ----
  zero_f32_kernel<<<cdiv((long)N_GRAPHS * HID, 256), 256, 0, stream>>>(pooled, N_GRAPHS * HID);
  pool_kernel<<<cdiv((long)N_NODES * (HID / 4), 256), 256, 0, stream>>>(bufB, batch, pooled);

  // ---- head MLP ----
  gemm_wmma_bf16_kernel<HID, HEAD_H><<<N_GRAPHS / 16, 256, 0, stream>>>(
      pooled, nullptr, Wph0, bh0, hidden, 1);
  gemm_wmma_bf16_kernel<HEAD_H, OUT_F><<<N_GRAPHS / 16, 256, 0, stream>>>(
      hidden, nullptr, Wph1, bh1, out, 0);

  (void)in_sizes; (void)n_in; (void)out_size; (void)ws_size;
}
